// GAB_32856499814496
// MI455X (gfx1250) — matrix operand order repaired and checked by reading, not compiled, not run on hardware
//
#include <hip/hip_runtime.h>

// ---------------------------------------------------------------------------
// CDNA5 (gfx1250) fused attention-pool pipeline, bf16 WMMA path.
//  - GEMM1/2/3: v_wmma_f32_16x16x32_bf16, A-strip staged in LDS via async
//    global->LDS copies (ASYNCcnt), 4 column-tiles per wave reuse one A frag.
//  - feature = sum_n features*score - x_query (softmax weights sum to 1),
//    so only the `features` intermediate is materialized (bf16, 128 MB).
// ---------------------------------------------------------------------------

typedef __attribute__((ext_vector_type(16))) __bf16 v16bf;
typedef __attribute__((ext_vector_type(8)))  float  v8f;
typedef __attribute__((ext_vector_type(4)))  int    v4i;

#if defined(__has_builtin)
#if __has_builtin(__builtin_amdgcn_global_load_async_to_lds_b128)
#define HAVE_ASYNC_B128 1
#endif
#if __has_builtin(__builtin_amdgcn_s_wait_asynccnt)
#define HAVE_WAIT_ASYNC 1
#endif
#endif

union FragU {
  v16bf v;
  unsigned short u[16];
  uint4 q[2];
};

__device__ __forceinline__ float bf2f(unsigned short h) {
  unsigned int x = ((unsigned int)h) << 16;
  float f;
  __builtin_memcpy(&f, &x, 4);
  return f;
}

__device__ __forceinline__ unsigned short f2bf(float f) {
  unsigned int x;
  __builtin_memcpy(&x, &f, 4);
  unsigned int r = x + 0x7FFFu + ((x >> 16) & 1u);  // round-to-nearest-even
  return (unsigned short)(r >> 16);
}

// ---------------------------------------------------------------------------
// Async global -> LDS 16B copy (gfx1250 GLOBAL_LOAD_ASYNC_TO_LDS_B128).
// Builtin signature (from hipcc diagnostic): (int4 AS1* src, int4 AS3* dst,
// imm offset, imm cpol). Falls back to load + ds_store if unavailable.
// ---------------------------------------------------------------------------
__device__ __forceinline__ void async_copy_b128(void* lds_dst, void* gsrc) {
#ifdef HAVE_ASYNC_B128
  __builtin_amdgcn_global_load_async_to_lds_b128(
      (__attribute__((address_space(1))) v4i*)gsrc,
      (__attribute__((address_space(3))) v4i*)lds_dst,
      0, 0);
#else
  *(uint4*)lds_dst = *(const uint4*)gsrc;
#endif
}

__device__ __forceinline__ void wait_async_done() {
#ifdef HAVE_ASYNC_B128
#ifdef HAVE_WAIT_ASYNC
  __builtin_amdgcn_s_wait_asynccnt(0);
#else
  asm volatile("s_wait_asynccnt 0" ::: "memory");
#endif
#endif
}

// ---------------------------------------------------------------------------
// f32 -> bf16 conversion, 8 elements / thread (b128 in, b128 out)
// ---------------------------------------------------------------------------
__global__ void cvt_f32_bf16(const float* __restrict__ in,
                             unsigned short* __restrict__ out, int n) {
  int i = (blockIdx.x * blockDim.x + threadIdx.x) << 3;
  if (i >= n) return;
  float4 a = *(const float4*)(in + i);
  float4 b = *(const float4*)(in + i + 4);
  union { unsigned short u[8]; uint4 q; } r;
  r.u[0] = f2bf(a.x); r.u[1] = f2bf(a.y); r.u[2] = f2bf(a.z); r.u[3] = f2bf(a.w);
  r.u[4] = f2bf(b.x); r.u[5] = f2bf(b.y); r.u[6] = f2bf(b.z); r.u[7] = f2bf(b.w);
  *(uint4*)(out + i) = r.q;
}

// ---------------------------------------------------------------------------
// WMMA fragment loaders.
// A 16x32 bf16 (from LDS strip [16][K]): lane = M row; lanes 0-15 hold
// K {0..7,16..23}, lanes 16-31 hold K {8..15,24..31} -> two 16B ds loads.
// B 32x16 bf16 (from row-major W [Nout,K]): lane = N column; lanes 0-15 hold
// K {0..7,16..23}, lanes 16-31 K {8..15,24..31} -> two 16B global loads.
// ---------------------------------------------------------------------------
__device__ __forceinline__ v16bf load_frag_A(const unsigned short* base,
                                             int lda, int lane) {
  int m  = lane & 15;
  int kb = (lane & 16) ? 8 : 0;
  const unsigned short* p = base + (size_t)m * lda + kb;
  FragU f;
  f.q[0] = *(const uint4*)(p);       // K kb .. kb+7
  f.q[1] = *(const uint4*)(p + 16);  // K kb+16 .. kb+23
  return f.v;
}

__device__ __forceinline__ v16bf load_frag_B(const unsigned short* __restrict__ Wk0,
                                             int ldw, int n0, int lane) {
  int n  = lane & 15;
  int kb = (lane & 16) ? 8 : 0;
  const unsigned short* p = Wk0 + (size_t)(n0 + n) * ldw + kb;
  FragU f;
  f.q[0] = *(const uint4*)(p);
  f.q[1] = *(const uint4*)(p + 16);
  return f.v;
}

// C/D tile epilogue: per-column BN (folded scale/bias) + LeakyReLU (+ add).
__device__ __forceinline__ void tile_epilogue(v8f c, unsigned short* __restrict__ out,
                                              int ldo, int row0, int col0, int lane,
                                              const float* __restrict__ gg,
                                              const float* __restrict__ bb,
                                              const float* __restrict__ mm,
                                              const float* __restrict__ vv,
                                              const float* __restrict__ addrow) {
  const int col   = col0 + (lane & 15);
  const int rbase = row0 + ((lane & 16) ? 8 : 0);
  const float scale = gg[col] * rsqrtf(vv[col] + 1e-5f);
  const float bias  = bb[col] - mm[col] * scale;
  const float addv  = addrow ? addrow[col] : 0.0f;
#pragma unroll
  for (int r = 0; r < 8; ++r) {
    float y = c[r] * scale + bias;
    y = (y >= 0.0f) ? y : 0.2f * y;
    out[(size_t)(rbase + r) * ldo + col] = f2bf(y + addv);
  }
}

// ---------------------------------------------------------------------------
// GEMM + BN + LeakyReLU (+ optional per-batch row-vector add)
//   out[M,Nout] (bf16) = act( A[M,K] @ W[Nout,K]^T ),  lda == ldw == K.
// Block: (blockDim/32) waves. The 16xK A strip for the block's row tile is
// async-staged into LDS once, then each wave computes a 16x64 tile (4
// accumulators sharing one A fragment per k-step).
// Dynamic LDS: 16*K*2 bytes.
// ---------------------------------------------------------------------------
__global__ void gemm_bn_lrelu(const unsigned short* __restrict__ A,
                              const unsigned short* __restrict__ W,
                              unsigned short* __restrict__ out, int ldo,
                              int K,
                              const float* __restrict__ gg, const float* __restrict__ bb,
                              const float* __restrict__ mm, const float* __restrict__ vv,
                              const float* __restrict__ addvec, int add_ld,
                              int rows_per_batch) {
  extern __shared__ unsigned short lds_a[];  // [16][K]
  const int lane = threadIdx.x & 31;
  const int wv   = threadIdx.x >> 5;
  const int row0 = blockIdx.x << 4;
  const int colw = (blockIdx.y * (blockDim.x >> 5) + wv) << 6;  // 64 cols/wave

  const unsigned short* Abase = A + (size_t)row0 * K;

  // Stage the 16xK bf16 A strip into LDS with async b128 copies.
  const int segs = (16 * K) >> 3;  // 16B segments
  for (int s = threadIdx.x; s < segs; s += blockDim.x)
    async_copy_b128(&lds_a[s << 3], (void*)(Abase + ((size_t)s << 3)));
  wait_async_done();
  __syncthreads();

  v8f acc0 = {}, acc1 = {}, acc2 = {}, acc3 = {};
  for (int k0 = 0; k0 < K; k0 += 32) {
    __builtin_prefetch(W + (size_t)colw * K + k0 + 256, 0, 3);
    v16bf a  = load_frag_A(lds_a + k0, K, lane);
    v16bf b0 = load_frag_B(W + k0, K, colw,      lane);
    v16bf b1 = load_frag_B(W + k0, K, colw + 16, lane);
    v16bf b2 = load_frag_B(W + k0, K, colw + 32, lane);
    v16bf b3 = load_frag_B(W + k0, K, colw + 48, lane);
    acc0 = __builtin_amdgcn_wmma_f32_16x16x32_bf16(false, a, false, b0, (short)0, acc0, false, false);
    acc1 = __builtin_amdgcn_wmma_f32_16x16x32_bf16(false, a, false, b1, (short)0, acc1, false, false);
    acc2 = __builtin_amdgcn_wmma_f32_16x16x32_bf16(false, a, false, b2, (short)0, acc2, false, false);
    acc3 = __builtin_amdgcn_wmma_f32_16x16x32_bf16(false, a, false, b3, (short)0, acc3, false, false);
  }

  const float* addrow =
      addvec ? addvec + (size_t)(row0 / rows_per_batch) * add_ld : nullptr;
  tile_epilogue(acc0, out, ldo, row0, colw,      lane, gg, bb, mm, vv, addrow);
  tile_epilogue(acc1, out, ldo, row0, colw + 16, lane, gg, bb, mm, vv, addrow);
  tile_epilogue(acc2, out, ldo, row0, colw + 32, lane, gg, bb, mm, vv, addrow);
  tile_epilogue(acc3, out, ldo, row0, colw + 48, lane, gg, bb, mm, vv, addrow);
}

// ---------------------------------------------------------------------------
// GEMM4 (K=64 -> scalar) + BN + LeakyReLU : one thread per row
// ---------------------------------------------------------------------------
__global__ void logits_kernel(const unsigned short* __restrict__ s2,
                              const float* __restrict__ W4,
                              const float* __restrict__ g4, const float* __restrict__ b4,
                              const float* __restrict__ m4, const float* __restrict__ v4,
                              float* __restrict__ logits, int M) {
  int i = blockIdx.x * blockDim.x + threadIdx.x;
  if (i >= M) return;
  const unsigned short* p = s2 + (size_t)i * 64;
  float acc = 0.f;
#pragma unroll
  for (int k = 0; k < 64; ++k) acc += bf2f(p[k]) * W4[k];
  float scale = g4[0] * rsqrtf(v4[0] + 1e-5f);
  float y = (acc - m4[0]) * scale + b4[0];
  logits[i] = (y >= 0.f) ? y : 0.2f * y;
}

// ---------------------------------------------------------------------------
// Softmax over N per batch (one block per batch)
// ---------------------------------------------------------------------------
__global__ void softmax_kernel(const float* __restrict__ logits,
                               float* __restrict__ scores, int N) {
  __shared__ float red[256];
  const int b = blockIdx.x;
  const float* L = logits + (size_t)b * N;

  float mx = -3.4e38f;
  for (int n = threadIdx.x; n < N; n += blockDim.x) mx = fmaxf(mx, L[n]);
  red[threadIdx.x] = mx;
  __syncthreads();
  for (int s = blockDim.x >> 1; s > 0; s >>= 1) {
    if (threadIdx.x < s) red[threadIdx.x] = fmaxf(red[threadIdx.x], red[threadIdx.x + s]);
    __syncthreads();
  }
  mx = red[0];
  __syncthreads();

  float sum = 0.f;
  for (int n = threadIdx.x; n < N; n += blockDim.x) sum += __expf(L[n] - mx);
  red[threadIdx.x] = sum;
  __syncthreads();
  for (int s = blockDim.x >> 1; s > 0; s >>= 1) {
    if (threadIdx.x < s) red[threadIdx.x] += red[threadIdx.x + s];
    __syncthreads();
  }
  const float inv = 1.0f / red[0];

  for (int n = threadIdx.x; n < N; n += blockDim.x)
    scores[(size_t)b * N + n] = __expf(L[n] - mx) * inv;
}

// ---------------------------------------------------------------------------
// Weighted pool over N, then subtract x_query.
// ---------------------------------------------------------------------------
__global__ void pool_kernel(const unsigned short* __restrict__ feat,
                            const float* __restrict__ scores,
                            const float* __restrict__ x_query,
                            float* __restrict__ out, int N, int E) {
  const int b = blockIdx.y;
  const int e = blockIdx.x * blockDim.x + threadIdx.x;
  const unsigned short* V = feat + (size_t)b * N * E + e;
  const float* S = scores + (size_t)b * N;
  float acc = 0.f;
  for (int n = 0; n < N; ++n) {
    __builtin_prefetch(V + (size_t)(n + 16) * E, 0, 3);
    acc += bf2f(V[(size_t)n * E]) * S[n];
  }
  out[(size_t)b * E + e] = acc - x_query[(size_t)b * E + e];
}

// ---------------------------------------------------------------------------
extern "C" void kernel_launch(void* const* d_in, const int* in_sizes, int n_in,
                              void* d_out, int out_size, void* d_ws, size_t ws_size,
                              hipStream_t stream) {
  (void)in_sizes; (void)n_in; (void)out_size; (void)ws_size;

  const float* x_query = (const float*)d_in[0];   // [16,2048]
  const float* x_key   = (const float*)d_in[1];   // [16,2048,512]
  const float* W1 = (const float*)d_in[2];        // [2048,512]
  const float* W2 = (const float*)d_in[3];        // [512,2048]
  const float* W3 = (const float*)d_in[4];        // [64,512]
  const float* W4 = (const float*)d_in[5];        // [1,64]
  const float* g1 = (const float*)d_in[6],  *b1 = (const float*)d_in[7];
  const float* m1 = (const float*)d_in[8],  *v1 = (const float*)d_in[9];
  const float* g2 = (const float*)d_in[10], *b2 = (const float*)d_in[11];
  const float* m2 = (const float*)d_in[12], *v2 = (const float*)d_in[13];
  const float* g3 = (const float*)d_in[14], *b3 = (const float*)d_in[15];
  const float* m3 = (const float*)d_in[16], *v3 = (const float*)d_in[17];
  const float* g4 = (const float*)d_in[18], *b4 = (const float*)d_in[19];
  const float* m4 = (const float*)d_in[20], *v4 = (const float*)d_in[21];

  const int B = 16, N = 2048, D = 512, E = 2048, K3 = 64;
  const int M = B * N;  // 32768 fused rows

  // Workspace layout (bf16 regions, all 16B-aligned), ~210 MB total.
  unsigned short* ws16  = (unsigned short*)d_ws;
  size_t off = 0;
  unsigned short* xkeyb = ws16 + off; off += (size_t)M * D;    // 32 MB
  unsigned short* W1b   = ws16 + off; off += (size_t)E * D;    // 2 MB
  unsigned short* W2b   = ws16 + off; off += (size_t)D * E;    // 2 MB
  unsigned short* W3b   = ws16 + off; off += (size_t)K3 * D;   // 64 KB
  unsigned short* feat  = ws16 + off; off += (size_t)M * E;    // 128 MB
  unsigned short* s1    = ws16 + off; off += (size_t)M * D;    // 32 MB
  unsigned short* s2    = ws16 + off; off += (size_t)M * K3;   // 4 MB
  float* logits = (float*)(ws16 + off);
  float* scores = logits + M;

  auto cdiv = [](int a, int b) { return (a + b - 1) / b; };

  // Stage bf16 copies of x_key and weights (b128 in / b128 out).
  cvt_f32_bf16<<<cdiv(M * D / 8, 256), 256, 0, stream>>>(x_key, xkeyb, M * D);
  cvt_f32_bf16<<<cdiv(E * D / 8, 256), 256, 0, stream>>>(W1, W1b, E * D);
  cvt_f32_bf16<<<cdiv(D * E / 8, 256), 256, 0, stream>>>(W2, W2b, D * E);
  cvt_f32_bf16<<<cdiv(K3 * D / 8, 256), 256, 0, stream>>>(W3, W3b, K3 * D);

  // GEMM1: features = lrelu(bn1(x_key @ W1^T)) + x_query   [M,2048]
  gemm_bn_lrelu<<<dim3(M / 16, E / 512), 256, 16 * D * 2, stream>>>(
      xkeyb, W1b, feat, E, D, g1, b1, m1, v1, x_query, E, N);

  // GEMM2: s1 = lrelu(bn2(features @ W2^T))                [M,512]
  gemm_bn_lrelu<<<dim3(M / 16, 1), 256, 16 * E * 2, stream>>>(
      feat, W2b, s1, D, E, g2, b2, m2, v2, nullptr, 0, 1);

  // GEMM3: s2 = lrelu(bn3(s1 @ W3^T))                      [M,64] (1 wave)
  gemm_bn_lrelu<<<dim3(M / 16, 1), 32, 16 * D * 2, stream>>>(
      s1, W3b, s2, K3, D, g3, b3, m3, v3, nullptr, 0, 1);

  // GEMM4 + BN4 + lrelu -> logits [M]
  logits_kernel<<<cdiv(M, 256), 256, 0, stream>>>(s2, W4, g4, b4, m4, v4, logits, M);

  // Softmax over N per batch -> scores [B,N]
  softmax_kernel<<<B, 256, 0, stream>>>(logits, scores, N);

  // Pool: out[b,e] = sum_n feat[b,n,e]*scores[b,n] - x_query[b,e]
  pool_kernel<<<dim3(E / 256, B), 256, 0, stream>>>(feat, scores, x_query,
                                                    (float*)d_out, N, E);
}
